// SymbolsDecoder_59012850647755
// MI455X (gfx1250) — matrix operand order repaired and checked by reading, not compiled, not run on hardware
//
#include <hip/hip_runtime.h>
#include <hip/hip_bf16.h>
#include <math.h>
#include <stdint.h>

// ---------------------------------------------------------------------------
// MI455X (gfx1250) decoder: bf16 WMMA GEMMs for all dense matmuls,
// VALU + wave32 shuffle reductions for the per-batch matvecs (which are
// L2-resident: enc 42MB + syms 105MB < 192MB global L2).
// Round 2: 2x2 output tiles per wave (32x32) in the big GEMMs -> 2x fewer
// fragment loads per WMMA; b128 loads in the logits matvec; prefetch hints.
// ---------------------------------------------------------------------------

typedef __bf16 bf16;
typedef __attribute__((ext_vector_type(16))) __bf16 v16bf;
typedef __attribute__((ext_vector_type(8)))  __bf16 v8bf;
typedef __attribute__((ext_vector_type(8)))  float  v8f;

#define BSZ   512
#define DDIM  256
#define LLEN  80
#define NSYM  200
#define TFULL 24
#define TSTEP 23
#define VOCAB 203

// ---------------- fragment loaders (CDNA5 ISA 7.12.2 wave32 layouts) -------
// A (16x32 bf16): lane&15 = row; lanes 0-15 hold K {0..7,16..23}, 16-31 {8..15,24..31}
__device__ __forceinline__ v16bf load_a_frag(const bf16* __restrict__ ap,
                                             int kk, int akb) {
  v8bf lo = *(const v8bf*)(ap + kk + akb);
  v8bf hi = *(const v8bf*)(ap + kk + 16 + akb);
  v16bf r;
  #pragma unroll
  for (int i = 0; i < 8; ++i) { r[i] = lo[i]; r[8 + i] = hi[i]; }
  return r;
}
// B (32x16 bf16): lane&15 = col; lanes 0-15 hold K {0..7,16..23}, 16-31 {8..15,24..31}
__device__ __forceinline__ v16bf load_b_frag(const bf16* __restrict__ bp,
                                             int kk, int bkb) {
  v8bf lo = *(const v8bf*)(bp + kk + bkb);
  v8bf hi = *(const v8bf*)(bp + kk + bkb + 16);
  v16bf r;
  #pragma unroll
  for (int i = 0; i < 8; ++i) { r[i] = lo[i]; r[8 + i] = hi[i]; }
  return r;
}

// ---------------- elementwise f32 -> bf16 weight conversion ----------------
__global__ void f32_to_bf16_kernel(const float* __restrict__ src,
                                   bf16* __restrict__ dst, int n) {
  int i = blockIdx.x * blockDim.x + threadIdx.x;
  if (i < n) dst[i] = (bf16)src[i];
}

// ---------------- target embedding gather (bf16 output) -------------------
__global__ void embed_kernel(const int* __restrict__ idxs,
                             const float* __restrict__ special,
                             const float* __restrict__ syms,
                             bf16* __restrict__ xbf) {
  int b = blockIdx.x, t = blockIdx.y, d = threadIdx.x;
  int idx = idxs[b * TFULL + t];
  float v;
  if (idx < 3) {
    int i2 = idx < 0 ? 0 : idx;
    v = special[i2 * DDIM + d];
  } else {
    int i2 = idx - 3;
    if (i2 > NSYM - 1) i2 = NSYM - 1;
    v = syms[((size_t)b * NSYM + i2) * DDIM + d];
  }
  xbf[((size_t)t * BSZ + b) * DDIM + d] = (bf16)v;
}

// ---------------- dual-pair bf16 WMMA GEMM, 1 tile / wave ------------------
// D[m,n] = sum_k A1[m,k]*B1[n,k] + sum_k A2[m,k]*B2[n,k] + bias1[n]+bias2[n]
// Used for N not divisible by 32 (attention scores, N=80).
__global__ void gemm2_wmma_kernel(
    const bf16* __restrict__ A1, int lda1, const bf16* __restrict__ B1, int ldb1, int K1,
    const bf16* __restrict__ A2, int lda2, const bf16* __restrict__ B2, int ldb2, int K2,
    const float* __restrict__ bias1, const float* __restrict__ bias2,
    float* __restrict__ outF, bf16* __restrict__ outB, int ldo, int Ntiles, int act)
{
  const int wave = threadIdx.x >> 5;
  const int lane = threadIdx.x & 31;
  const int tile = blockIdx.x * (blockDim.x >> 5) + wave;
  const int tm = tile / Ntiles, tn = tile % Ntiles;
  const int m0 = tm * 16, n0 = tn * 16;

  const int arow = m0 + (lane & 15);
  const int akb  = (lane >> 4) * 8;
  const int bcol = n0 + (lane & 15);
  const int bkb  = (lane >> 4) * 8;

  v8f acc = {};
  #pragma unroll
  for (int pass = 0; pass < 2; ++pass) {
    const bf16* A  = pass ? A2 : A1;
    const bf16* Bw = pass ? B2 : B1;
    const int lda  = pass ? lda2 : lda1;
    const int ldb  = pass ? ldb2 : ldb1;
    const int K    = pass ? K2 : K1;
    if (K == 0) continue;
    const bf16* ap = A  + (size_t)arow * lda;
    const bf16* bp = Bw + (size_t)bcol * ldb;
    if (pass == 0 && K2 > 0) {
      // warm the second-pass lines while pass 0 computes
      __builtin_prefetch(A2 + (size_t)arow * lda2, 0, 1);
      __builtin_prefetch(B2 + (size_t)bcol * ldb2, 0, 1);
    }
    for (int kk = 0; kk < K; kk += 32) {
      v16bf a = load_a_frag(ap, kk, akb);
      v16bf b = load_b_frag(bp, kk, bkb);
      acc = __builtin_amdgcn_wmma_f32_16x16x32_bf16(
          false, a, false, b, (short)0, acc, false, false);
    }
  }

  float bias = 0.f;
  if (bias1) bias += bias1[bcol];
  if (bias2) bias += bias2[bcol];

  const int mbase = m0 + (lane >> 4) * 8;
  #pragma unroll
  for (int r = 0; r < 8; ++r) {
    float v = acc[r] + bias;
    if (act == 1) v = v > 0.f ? v : 0.f;
    size_t o = (size_t)(mbase + r) * ldo + bcol;
    if (outF) outF[o] = v;
    if (outB) outB[o] = (bf16)v;
  }
}

// ---------------- dual-pair bf16 WMMA GEMM, 2x2 tiles / wave ---------------
// Each wave owns a 32x32 output block: A fragments reused across two N-tiles,
// B fragments reused across two M-tiles -> 4 WMMAs per 4 fragment loads.
__global__ void gemm2_wmma_2x2_kernel(
    const bf16* __restrict__ A1, int lda1, const bf16* __restrict__ B1, int ldb1, int K1,
    const bf16* __restrict__ A2, int lda2, const bf16* __restrict__ B2, int ldb2, int K2,
    const float* __restrict__ bias1, const float* __restrict__ bias2,
    float* __restrict__ outF, bf16* __restrict__ outB, int ldo, int Ntiles2, int act)
{
  const int wave = threadIdx.x >> 5;
  const int lane = threadIdx.x & 31;
  const int tile = blockIdx.x * (blockDim.x >> 5) + wave;
  const int tm = tile / Ntiles2, tn = tile % Ntiles2;
  const int m0 = tm * 32, n0 = tn * 32;

  const int arow = lane & 15;
  const int akb  = (lane >> 4) * 8;
  const int bcol = lane & 15;
  const int bkb  = (lane >> 4) * 8;

  v8f acc00 = {}, acc01 = {}, acc10 = {}, acc11 = {};

  #pragma unroll
  for (int pass = 0; pass < 2; ++pass) {
    const bf16* A  = pass ? A2 : A1;
    const bf16* Bw = pass ? B2 : B1;
    const int lda  = pass ? lda2 : lda1;
    const int ldb  = pass ? ldb2 : ldb1;
    const int K    = pass ? K2 : K1;
    if (K == 0) continue;
    const bf16* ap0 = A  + (size_t)(m0 + arow) * lda;
    const bf16* ap1 = A  + (size_t)(m0 + 16 + arow) * lda;
    const bf16* bp0 = Bw + (size_t)(n0 + bcol) * ldb;
    const bf16* bp1 = Bw + (size_t)(n0 + 16 + bcol) * ldb;
    if (pass == 0 && K2 > 0) {
      __builtin_prefetch(A2 + (size_t)(m0 + arow) * lda2, 0, 1);
      __builtin_prefetch(A2 + (size_t)(m0 + 16 + arow) * lda2, 0, 1);
      __builtin_prefetch(B2 + (size_t)(n0 + bcol) * ldb2, 0, 1);
      __builtin_prefetch(B2 + (size_t)(n0 + 16 + bcol) * ldb2, 0, 1);
    }
    for (int kk = 0; kk < K; kk += 32) {
      v16bf a0 = load_a_frag(ap0, kk, akb);
      v16bf a1 = load_a_frag(ap1, kk, akb);
      v16bf b0 = load_b_frag(bp0, kk, bkb);
      v16bf b1 = load_b_frag(bp1, kk, bkb);
      acc00 = __builtin_amdgcn_wmma_f32_16x16x32_bf16(
          false, a0, false, b0, (short)0, acc00, false, false);
      acc01 = __builtin_amdgcn_wmma_f32_16x16x32_bf16(
          false, a0, false, b1, (short)0, acc01, false, false);
      acc10 = __builtin_amdgcn_wmma_f32_16x16x32_bf16(
          false, a1, false, b0, (short)0, acc10, false, false);
      acc11 = __builtin_amdgcn_wmma_f32_16x16x32_bf16(
          false, a1, false, b1, (short)0, acc11, false, false);
    }
  }

  const int nc0 = n0 + bcol;        // columns of acc*0
  const int nc1 = n0 + 16 + bcol;   // columns of acc*1
  float bias_0 = 0.f, bias_1 = 0.f;
  if (bias1) { bias_0 += bias1[nc0]; bias_1 += bias1[nc1]; }
  if (bias2) { bias_0 += bias2[nc0]; bias_1 += bias2[nc1]; }

  const int mb0 = m0 + (lane >> 4) * 8;        // rows of acc0*
  const int mb1 = m0 + 16 + (lane >> 4) * 8;   // rows of acc1*
  #pragma unroll
  for (int r = 0; r < 8; ++r) {
    float v00 = acc00[r] + bias_0;
    float v01 = acc01[r] + bias_1;
    float v10 = acc10[r] + bias_0;
    float v11 = acc11[r] + bias_1;
    if (act == 1) {
      v00 = v00 > 0.f ? v00 : 0.f;
      v01 = v01 > 0.f ? v01 : 0.f;
      v10 = v10 > 0.f ? v10 : 0.f;
      v11 = v11 > 0.f ? v11 : 0.f;
    }
    size_t o00 = (size_t)(mb0 + r) * ldo + nc0;
    size_t o01 = (size_t)(mb0 + r) * ldo + nc1;
    size_t o10 = (size_t)(mb1 + r) * ldo + nc0;
    size_t o11 = (size_t)(mb1 + r) * ldo + nc1;
    if (outF) { outF[o00] = v00; outF[o01] = v01; outF[o10] = v10; outF[o11] = v11; }
    if (outB) {
      outB[o00] = (bf16)v00; outB[o01] = (bf16)v01;
      outB[o10] = (bf16)v10; outB[o11] = (bf16)v11;
    }
  }
}

// ---------------- softmax over L=80 + context matvec -----------------------
// One block (256 threads) per batch element. ctx[b,d] = sum_l p[l]*enc[b,l,d].
__global__ void softmax_ctx_kernel(const float* __restrict__ aw,
                                   const uint8_t* __restrict__ emask,
                                   const float* __restrict__ enc,
                                   bf16* __restrict__ ctx)
{
  __shared__ float p[LLEN];
  __shared__ float red[256];
  const int b = blockIdx.x, t = threadIdx.x;

  float v = -3.0e38f;
  if (t < LLEN) {
    v = aw[b * LLEN + t];
    if (!emask[b * LLEN + t]) v = -3.0e38f;
    p[t] = v;
  }
  red[t] = v;
  __syncthreads();
  for (int s = 128; s > 0; s >>= 1) {
    if (t < s) red[t] = fmaxf(red[t], red[t + s]);
    __syncthreads();
  }
  const float mx = red[0];
  __syncthreads();
  float e = 0.f;
  if (t < LLEN) { e = __expf(p[t] - mx); p[t] = e; }
  red[t] = e;
  __syncthreads();
  for (int s = 128; s > 0; s >>= 1) {
    if (t < s) red[t] += red[t + s];
    __syncthreads();
  }
  const float inv = 1.f / red[0];
  __syncthreads();
  if (t < LLEN) p[t] *= inv;
  __syncthreads();

  const float* eb = enc + (size_t)b * LLEN * DDIM;
  float acc = 0.f;
  #pragma unroll 4
  for (int l = 0; l < LLEN; ++l) acc = fmaf(p[l], eb[l * DDIM + t], acc);
  ctx[(size_t)b * DDIM + t] = (bf16)acc;
}

// ---------------- LSTM pointwise cell update -------------------------------
// gates [B,4D] in PyTorch order i,f,g,o. Updates c in place, writes bf16 h.
__global__ void lstm_point_kernel(const float* __restrict__ gates,
                                  float* __restrict__ c,
                                  bf16* __restrict__ hbf)
{
  const int i = blockIdx.x * blockDim.x + threadIdx.x;  // b*D + d
  const int b = i >> 8, d = i & 255;
  const float* g = gates + (size_t)b * 4 * DDIM;
  const float gi = g[d];
  const float gf = g[DDIM + d];
  const float gg = g[2 * DDIM + d];
  const float go = g[3 * DDIM + d];
  const float si = 1.f / (1.f + __expf(-gi));
  const float sf = 1.f / (1.f + __expf(-gf));
  const float so = 1.f / (1.f + __expf(-go));
  const float c2 = sf * c[i] + si * tanhf(gg);
  c[i] = c2;
  hbf[i] = (bf16)(so * tanhf(c2));
}

// ---------------- vocab logits + log_softmax -------------------------------
// One block (8 waves) per batch element. Per-row dot via coalesced b128
// loads + wave32 shuffle reduction (per-batch matrices, so no WMMA mapping).
__global__ void logits_kernel(const float* __restrict__ outv,
                              const float* __restrict__ syms,
                              const float* __restrict__ special,
                              const uint8_t* __restrict__ smask,
                              float* __restrict__ dout, int t)
{
  __shared__ float so_[DDIM];
  __shared__ float lg[VOCAB];
  __shared__ float red[256];
  const int b = blockIdx.x;
  const int tid = threadIdx.x;
  const int wave = tid >> 5, lane = tid & 31;

  so_[tid] = outv[(size_t)b * DDIM + tid];
  __syncthreads();
  const float4* so4 = (const float4*)so_;

  for (int r = wave; r < VOCAB; r += 8) {
    const float* row = (r < 3) ? (special + r * DDIM)
                               : (syms + ((size_t)b * NSYM + (r - 3)) * DDIM);
    const float4* row4 = (const float4*)row;
    float s = 0.f;
    #pragma unroll
    for (int j = 0; j < DDIM / (4 * 32); ++j) {
      float4 rv = row4[lane + j * 32];
      float4 ov = so4[lane + j * 32];
      s = fmaf(rv.x, ov.x, s);
      s = fmaf(rv.y, ov.y, s);
      s = fmaf(rv.z, ov.z, s);
      s = fmaf(rv.w, ov.w, s);
    }
    #pragma unroll
    for (int off = 16; off > 0; off >>= 1) s += __shfl_xor(s, off, 32);
    if (lane == 0) {
      if (r >= 3 && !smask[b * NSYM + (r - 3)]) s = -3.0e38f;
      lg[r] = s;
    }
  }
  __syncthreads();

  const float v = (tid < VOCAB) ? lg[tid] : -3.0e38f;
  red[tid] = v;
  __syncthreads();
  for (int s = 128; s > 0; s >>= 1) {
    if (tid < s) red[tid] = fmaxf(red[tid], red[tid + s]);
    __syncthreads();
  }
  const float mx = red[0];
  __syncthreads();
  red[tid] = (tid < VOCAB) ? __expf(v - mx) : 0.f;
  __syncthreads();
  for (int s = 128; s > 0; s >>= 1) {
    if (tid < s) red[tid] += red[tid + s];
    __syncthreads();
  }
  const float lse = __logf(red[0]);
  if (tid < VOCAB)
    dout[((size_t)b * TSTEP + t) * VOCAB + tid] = v - mx - lse;
}

// ===========================================================================
extern "C" void kernel_launch(void* const* d_in, const int* in_sizes, int n_in,
                              void* d_out, int out_size, void* d_ws, size_t ws_size,
                              hipStream_t stream)
{
  (void)in_sizes; (void)n_in; (void)out_size; (void)ws_size;

  const float*   enc     = (const float*)d_in[0];
  const float*   syms    = (const float*)d_in[1];
  const int*     tidx    = (const int*)d_in[2];
  const uint8_t* emask   = (const uint8_t*)d_in[3];
  const uint8_t* smask   = (const uint8_t*)d_in[4];
  const float*   special = (const float*)d_in[5];
  const float*   attnW   = (const float*)d_in[6];
  const float*   attnB   = (const float*)d_in[7];
  const float*   combW   = (const float*)d_in[8];
  const float*   combB   = (const float*)d_in[9];
  const float*   outW    = (const float*)d_in[10];
  const float*   outBias = (const float*)d_in[11];
  const float*   Wih0    = (const float*)d_in[12];
  const float*   Whh0    = (const float*)d_in[13];
  const float*   bih0    = (const float*)d_in[14];
  const float*   bhh0    = (const float*)d_in[15];
  const float*   Wih1    = (const float*)d_in[16];
  const float*   Whh1    = (const float*)d_in[17];
  const float*   bih1    = (const float*)d_in[18];
  const float*   bhh1    = (const float*)d_in[19];

  float* dout = (float*)d_out;

  // ---- workspace carve-out (256B aligned slabs) ----
  char* ws = (char*)d_ws;
  size_t off = 0;
  auto alloc = [&](size_t bytes) -> char* {
    char* p = ws + off;
    off = (off + bytes + 255) & ~(size_t)255;
    return p;
  };
  bf16* attnW_b = (bf16*)alloc((size_t)LLEN * 512 * 2);
  bf16* combW_b = (bf16*)alloc((size_t)DDIM * 512 * 2);
  bf16* Wih0_b  = (bf16*)alloc((size_t)4 * DDIM * DDIM * 2);
  bf16* Whh0_b  = (bf16*)alloc((size_t)4 * DDIM * DDIM * 2);
  bf16* Wih1_b  = (bf16*)alloc((size_t)4 * DDIM * DDIM * 2);
  bf16* Whh1_b  = (bf16*)alloc((size_t)4 * DDIM * DDIM * 2);
  bf16* outW_b  = (bf16*)alloc((size_t)DDIM * DDIM * 2);
  bf16* x_b     = (bf16*)alloc((size_t)TSTEP * BSZ * DDIM * 2);
  bf16* h0_b    = (bf16*)alloc((size_t)BSZ * DDIM * 2);
  bf16* h1_b    = (bf16*)alloc((size_t)BSZ * DDIM * 2);
  bf16* ctx_b   = (bf16*)alloc((size_t)BSZ * DDIM * 2);
  bf16* rnn_b   = (bf16*)alloc((size_t)BSZ * DDIM * 2);
  float* c0     = (float*)alloc((size_t)BSZ * DDIM * 4);
  float* c1     = (float*)alloc((size_t)BSZ * DDIM * 4);
  float* aw     = (float*)alloc((size_t)BSZ * LLEN * 4);
  float* gates  = (float*)alloc((size_t)BSZ * 4 * DDIM * 4);
  float* outF   = (float*)alloc((size_t)BSZ * DDIM * 4);

  // ---- weight conversion to bf16 ----
  auto cvt = [&](const float* s, bf16* d, int n) {
    f32_to_bf16_kernel<<<(n + 255) / 256, 256, 0, stream>>>(s, d, n);
  };
  cvt(attnW, attnW_b, LLEN * 512);
  cvt(combW, combW_b, DDIM * 512);
  cvt(Wih0,  Wih0_b,  4 * DDIM * DDIM);
  cvt(Whh0,  Whh0_b,  4 * DDIM * DDIM);
  cvt(Wih1,  Wih1_b,  4 * DDIM * DDIM);
  cvt(Whh1,  Whh1_b,  4 * DDIM * DDIM);
  cvt(outW,  outW_b,  DDIM * DDIM);

  // ---- target embeddings (all timesteps at once) ----
  embed_kernel<<<dim3(BSZ, TSTEP), DDIM, 0, stream>>>(tidx, special, syms, x_b);

  // ---- zero initial state ----
  hipMemsetAsync(h0_b, 0, (size_t)BSZ * DDIM * 2, stream);
  hipMemsetAsync(h1_b, 0, (size_t)BSZ * DDIM * 2, stream);
  hipMemsetAsync(c0,   0, (size_t)BSZ * DDIM * 4, stream);
  hipMemsetAsync(c1,   0, (size_t)BSZ * DDIM * 4, stream);

  // ---- sequential decode ----
  for (int t = 0; t < TSTEP; ++t) {
    const bf16* xt = x_b + (size_t)t * BSZ * DDIM;

    // attention scores: [x_t, h1] @ attn_W.T  -> aw [B,80]
    // (N=80 not divisible by 32 -> 1-tile kernel; 160 tiles = 20 blocks)
    gemm2_wmma_kernel<<<20, 256, 0, stream>>>(
        xt, DDIM, attnW_b, 512, DDIM,
        h1_b, DDIM, attnW_b + DDIM, 512, DDIM,
        attnB, nullptr, aw, nullptr, LLEN, LLEN / 16, 0);

    // softmax over L + context matvec (L2-resident encoder_outputs)
    softmax_ctx_kernel<<<BSZ, 256, 0, stream>>>(aw, emask, enc, ctx_b);

    // rnn_in = relu([x_t, ctx] @ comb_W.T) -> bf16 [B,256]; 128 waves
    gemm2_wmma_2x2_kernel<<<16, 256, 0, stream>>>(
        xt, DDIM, combW_b, 512, DDIM,
        ctx_b, DDIM, combW_b + DDIM, 512, DDIM,
        combB, nullptr, nullptr, rnn_b, DDIM, DDIM / 32, 1);

    // LSTM layer 0 gates: rnn_in@Wih0.T + h0@Whh0.T  [B,1024]; 512 waves
    gemm2_wmma_2x2_kernel<<<64, 256, 0, stream>>>(
        rnn_b, DDIM, Wih0_b, DDIM, DDIM,
        h0_b,  DDIM, Whh0_b, DDIM, DDIM,
        bih0, bhh0, gates, nullptr, 4 * DDIM, (4 * DDIM) / 32, 0);
    lstm_point_kernel<<<(BSZ * DDIM) / 256, 256, 0, stream>>>(gates, c0, h0_b);

    // LSTM layer 1 gates: h0n@Wih1.T + h1@Whh1.T
    gemm2_wmma_2x2_kernel<<<64, 256, 0, stream>>>(
        h0_b, DDIM, Wih1_b, DDIM, DDIM,
        h1_b, DDIM, Whh1_b, DDIM, DDIM,
        bih1, bhh1, gates, nullptr, 4 * DDIM, (4 * DDIM) / 32, 0);
    lstm_point_kernel<<<(BSZ * DDIM) / 256, 256, 0, stream>>>(gates, c1, h1_b);

    // out = h1n @ out_W.T + out_b  [B,256]; 128 waves
    gemm2_wmma_2x2_kernel<<<16, 256, 0, stream>>>(
        h1_b, DDIM, outW_b, DDIM, DDIM,
        nullptr, 0, nullptr, 0, 0,
        outBias, nullptr, outF, nullptr, DDIM, DDIM / 32, 0);

    // vocab projection + log_softmax (L2-resident symbols_encodings)
    logits_kernel<<<BSZ, 256, 0, stream>>>(outF, syms, special, smask, dout, t);
  }
}
